// KGAT_11269994185391
// MI455X (gfx1250) — compile-verified, matrix operand order checked by reading, not run on hardware
//
#include <hip/hip_runtime.h>
#include <hip/hip_bf16.h>
#include <math.h>

typedef __attribute__((ext_vector_type(16))) _Float16 v16h;
typedef __attribute__((ext_vector_type(8)))  _Float16 v8h;
typedef __attribute__((ext_vector_type(8)))  float    v8f;

#define D_DIM 64
#define NEG_SLOPE 0.01f
#define L2_EPS 1e-12f

// ISA 7.12.2: 16-bit 16x32 fragment for lane (mn, half) within a 32-wide K
// chunk is two contiguous 8-half runs: [kb+8*half, +8) and [kb+8*half+16, +8).
// A is row-major (M x 64); B is stored transposed (N x 64) so the same loader
// serves both operands as two 16-byte global loads.
__device__ __forceinline__ v16h load_frag(const _Float16* __restrict__ rowBase,
                                          int kb, int half) {
    v8h lo = *(const v8h*)(rowBase + kb + 8 * half);
    v8h hi = *(const v8h*)(rowBase + kb + 16 + 8 * half);
    return __builtin_shufflevector(lo, hi, 0, 1, 2, 3, 4, 5, 6, 7,
                                   8, 9, 10, 11, 12, 13, 14, 15);
}

// ---------- init / utility ----------
__global__ void k_init(float* nmax, float* nsum, float* hnb, int N) {
    long i = (long)blockIdx.x * blockDim.x + threadIdx.x;
    if (i < N) { nmax[i] = -INFINITY; nsum[i] = 0.f; }
    if (i < (long)N * D_DIM) hnb[i] = 0.f;
}

__global__ void k_zero(float* p, long n) {
    long i = (long)blockIdx.x * blockDim.x + threadIdx.x;
    if (i < n) p[i] = 0.f;
}

__global__ void k_f32_to_f16(const float* __restrict__ s, _Float16* __restrict__ d, long n) {
    long i = (long)blockIdx.x * blockDim.x + threadIdx.x;
    if (i < n) d[i] = (_Float16)s[i];
}

// pf16 = (f16)(a .* b)  -- fused elementwise product feeding the layer GEMMs
__global__ void k_prod_f16(const float* __restrict__ a, const float* __restrict__ b,
                           _Float16* __restrict__ d, long n) {
    long i = (long)blockIdx.x * blockDim.x + threadIdx.x;
    if (i < n) d[i] = (_Float16)(a[i] * b[i]);
}

// dstT[m][c][r] = (f16)src[m][r][c]   (batched small-matrix transpose to f16)
__global__ void k_transpose_f16(const float* __restrict__ src, _Float16* __restrict__ dst,
                                int rows, int cols, long total) {
    long i = (long)blockIdx.x * blockDim.x + threadIdx.x;
    if (i >= total) return;
    long rc = (long)rows * cols;
    long m = i / rc, rem = i % rc;
    int r = (int)(rem / cols), c = (int)(rem % cols);
    dst[m * rc + (long)c * rows + r] = (_Float16)src[i];
}

// ---------- T[r,n,k] = sum_d h[n,d] * W_R[r,d,k]  (WMMA f16->f32, store f16) ----------
// A: hf16 [N x 64] row-major; B: WRt [R][64 x 64] transposed (k-major rows).
__global__ void k_transform(const _Float16* __restrict__ hf16,
                            const _Float16* __restrict__ WRt,
                            _Float16* __restrict__ T, int N) {
    const int r = blockIdx.y;
    const int nodeBase = blockIdx.x * 16;
    const int tid = threadIdx.x;          // 128 threads = 4 waves -> 64 cols
    const int lane = tid & 31, wave = tid >> 5;
    const int half = lane >> 4, mn = lane & 15;
    const int colBase = wave * 16;

    int arow = nodeBase + mn; if (arow >= N) arow = N - 1;
    const _Float16* A = hf16 + (long)arow * 64;
    const _Float16* B = WRt + (long)r * 64 * 64 + (long)(colBase + mn) * 64;

    v8f c = {};
#pragma unroll
    for (int kb = 0; kb < 64; kb += 32) {
        v16h a = load_frag(A, kb, half);
        v16h b = load_frag(B, kb, half);
        c = __builtin_amdgcn_wmma_f32_16x16x32_f16(false, a, false, b, (short)0, c,
                                                   false, false);
    }
    _Float16* o = T + ((long)r * N + nodeBase) * 64 + colBase + mn;
    if (nodeBase + 16 <= N) {             // wave-uniform fast path
#pragma unroll
        for (int i = 0; i < 8; ++i) o[(long)(i + 8 * half) * 64] = (_Float16)c[i];
    } else {
#pragma unroll
        for (int i = 0; i < 8; ++i)
            if (nodeBase + i + 8 * half < N) o[(long)(i + 8 * half) * 64] = (_Float16)c[i];
    }
}

// ---------- y = leaky_relu(P @ W), P = (h .* h_nb) in f16, Wt transposed ----------
// Launch with blockDim.x = 2*NC (NC/16 waves), so every wave has work.
__global__ void k_layer(const _Float16* __restrict__ P,
                        const _Float16* __restrict__ Wt,
                        float* __restrict__ out, int N, int NC) {
    const int nodeBase = blockIdx.x * 16;
    const int tid = threadIdx.x;
    const int lane = tid & 31, wave = tid >> 5;
    const int half = lane >> 4, mn = lane & 15;
    const int colBase = wave * 16;

    int arow = nodeBase + mn; if (arow >= N) arow = N - 1;
    const _Float16* A = P + (long)arow * 64;
    const _Float16* B = Wt + (long)(colBase + mn) * 64;

    v8f c = {};
#pragma unroll
    for (int kb = 0; kb < 64; kb += 32) {
        v16h a = load_frag(A, kb, half);
        v16h b = load_frag(B, kb, half);
        c = __builtin_amdgcn_wmma_f32_16x16x32_f16(false, a, false, b, (short)0, c,
                                                   false, false);
    }
    float* o = out + (long)nodeBase * NC + colBase + mn;
    if (nodeBase + 16 <= N) {
#pragma unroll
        for (int i = 0; i < 8; ++i) {
            float v = c[i];
            o[(long)(i + 8 * half) * NC] = (v >= 0.f) ? v : NEG_SLOPE * v;
        }
    } else {
#pragma unroll
        for (int i = 0; i < 8; ++i) {
            if (nodeBase + i + 8 * half < N) {
                float v = c[i];
                o[(long)(i + 8 * half) * NC] = (v >= 0.f) ? v : NEG_SLOPE * v;
            }
        }
    }
}

// ---------- att[e] = sum_k T[r,src,k] * tanh(T[r,dst,k] + rel[r,k]); atomic max ----------
__global__ void k_attention(const _Float16* __restrict__ T,
                            const float* __restrict__ rel,
                            const int* __restrict__ src, const int* __restrict__ dst,
                            const int* __restrict__ ety,
                            float* __restrict__ attbuf, float* __restrict__ nmax,
                            int N, int E) {
    long gid = (long)blockIdx.x * blockDim.x + threadIdx.x;
    int g = (int)(gid & 7);               // 8 lanes/edge, 16B f16 loads
    long e = gid >> 3;
    if (e >= E) return;
    int s = src[e], d = dst[e], r = ety[e];
    v8h tv = *(const v8h*)(T + ((long)r * N + s) * 64 + g * 8);
    v8h hv = *(const v8h*)(T + ((long)r * N + d) * 64 + g * 8);
    const float* rr = rel + (long)r * 64 + g * 8;
    float partial = 0.f;
#pragma unroll
    for (int i = 0; i < 8; ++i)
        partial += (float)tv[i] * tanhf((float)hv[i] + rr[i]);
#pragma unroll
    for (int m = 1; m < 8; m <<= 1) partial += __shfl_xor(partial, m, 32);
    if (g == 0) {
        attbuf[e] = partial;
        if (partial >= 0.f) atomicMax((int*)(nmax + d), __float_as_int(partial));
        else                atomicMin((unsigned int*)(nmax + d), __float_as_uint(partial));
    }
}

// ---------- w[e] = exp(att - m[dst]); atomic sum per dst ----------
__global__ void k_expsum(const float* __restrict__ attbuf, const int* __restrict__ dst,
                         const float* __restrict__ nmax,
                         float* __restrict__ wbuf, float* __restrict__ nsum, int E) {
    long e = (long)blockIdx.x * blockDim.x + threadIdx.x;
    if (e >= E) return;
    int d = dst[e];
    float m = nmax[d];
    if (!(m > -1e37f)) m = 0.f;           // isfinite replacement (init was -inf)
    float v = expf(attbuf[e] - m);
    wbuf[e] = v;
    atomicAdd(nsum + d, v);
}

// ---------- h_nb[dst] += h[src] * (w_e / max(sum,eps)) ----------
__global__ void k_aggregate(const float* __restrict__ srcH,
                            const float* __restrict__ wbuf, const float* __restrict__ nsum,
                            const int* __restrict__ src, const int* __restrict__ dst,
                            float* __restrict__ hnb, int E) {
    long gid = (long)blockIdx.x * blockDim.x + threadIdx.x;
    int g = (int)(gid & 15);              // 16 lanes/edge x float4 = 256B row
    long e = gid >> 4;
    if (e >= E) return;
    int s = src[e], d = dst[e];
    float coef = wbuf[e] / fmaxf(nsum[d], L2_EPS);
    float4 hv = *(const float4*)(srcH + (long)s * 64 + g * 4);
    float* o = hnb + (long)d * 64 + g * 4;
    atomicAdd(o + 0, hv.x * coef);
    atomicAdd(o + 1, hv.y * coef);
    atomicAdd(o + 2, hv.z * coef);
    atomicAdd(o + 3, hv.w * coef);
}

// ---------- out = concat(h0, l2norm(h1), l2norm(h2)) ----------
__global__ void k_output(const float* __restrict__ h0, const float* __restrict__ h1,
                         const float* __restrict__ h2, float* __restrict__ out, int N) {
    int node = blockIdx.x * 8 + (threadIdx.x >> 5);
    int lane = threadIdx.x & 31;
    if (node >= N) return;
    const long ob = (long)node * 160;
    out[ob + lane]      = h0[(long)node * 64 + lane];
    out[ob + 32 + lane] = h0[(long)node * 64 + 32 + lane];
    float a0 = h1[(long)node * 64 + lane];
    float a1 = h1[(long)node * 64 + 32 + lane];
    float ss = a0 * a0 + a1 * a1;
#pragma unroll
    for (int m = 1; m < 32; m <<= 1) ss += __shfl_xor(ss, m, 32);
    float inv = 1.f / fmaxf(sqrtf(ss), L2_EPS);
    out[ob + 64 + lane] = a0 * inv;
    out[ob + 96 + lane] = a1 * inv;
    float b0 = h2[(long)node * 32 + lane];
    float s2 = b0 * b0;
#pragma unroll
    for (int m = 1; m < 32; m <<= 1) s2 += __shfl_xor(s2, m, 32);
    out[ob + 128 + lane] = b0 * (1.f / fmaxf(sqrtf(s2), L2_EPS));
}

extern "C" void kernel_launch(void* const* d_in, const int* in_sizes, int n_in,
                              void* d_out, int out_size, void* d_ws, size_t ws_size,
                              hipStream_t stream) {
    (void)n_in; (void)out_size; (void)ws_size;
    const float* ent = (const float*)d_in[0];
    const float* rel = (const float*)d_in[1];
    const float* WR  = (const float*)d_in[2];
    const float* W0  = (const float*)d_in[3];
    const float* W1  = (const float*)d_in[4];
    const int* src = (const int*)d_in[5];
    const int* dst = (const int*)d_in[6];
    const int* ety = (const int*)d_in[7];
    const int N = in_sizes[0] / D_DIM;
    const int R = in_sizes[1] / D_DIM;
    const int E = in_sizes[5];
    float* out = (float*)d_out;

    char* ws = (char*)d_ws;
    size_t off = 0;
    auto take = [&](size_t bytes) -> void* {
        void* p = ws + off;
        off = (off + bytes + 255) & ~(size_t)255;
        return p;
    };
    _Float16* T    = (_Float16*)take((size_t)R * N * D_DIM * sizeof(_Float16));
    _Float16* hf16 = (_Float16*)take((size_t)N * D_DIM * sizeof(_Float16));
    _Float16* pf16 = (_Float16*)take((size_t)N * D_DIM * sizeof(_Float16));
    _Float16* WRt  = (_Float16*)take((size_t)R * 64 * 64 * sizeof(_Float16));
    _Float16* W0t  = (_Float16*)take((size_t)64 * 64 * sizeof(_Float16));
    _Float16* W1t  = (_Float16*)take((size_t)64 * 32 * sizeof(_Float16));
    float* attbuf  = (float*)take((size_t)E * sizeof(float));
    float* wbuf    = (float*)take((size_t)E * sizeof(float));
    float* nmax    = (float*)take((size_t)N * sizeof(float));
    float* nsum    = (float*)take((size_t)N * sizeof(float));
    float* hnb     = (float*)take((size_t)N * D_DIM * sizeof(float));
    float* h1      = (float*)take((size_t)N * D_DIM * sizeof(float));
    float* h2      = (float*)take((size_t)N * 32 * sizeof(float));

    const long nd = (long)N * D_DIM;
    const dim3 b256(256);
    const unsigned ndBlocks = (unsigned)((nd + 255) / 256);
    const int nTiles = (N + 15) / 16;

    // prep
    k_init<<<dim3(ndBlocks), b256, 0, stream>>>(nmax, nsum, hnb, N);
    k_f32_to_f16<<<dim3(ndBlocks), b256, 0, stream>>>(ent, hf16, nd);
    {
        long t = (long)R * 64 * 64;
        k_transpose_f16<<<dim3((unsigned)((t + 255) / 256)), b256, 0, stream>>>(
            WR, WRt, 64, 64, t);
        k_transpose_f16<<<dim3((unsigned)((64 * 64 + 255) / 256)), b256, 0, stream>>>(
            W0, W0t, 64, 64, 64 * 64);
        k_transpose_f16<<<dim3((unsigned)((64 * 32 + 255) / 256)), b256, 0, stream>>>(
            W1, W1t, 64, 32, 64 * 32);
    }
    // per-relation transform (WMMA)
    k_transform<<<dim3(nTiles, R), dim3(128), 0, stream>>>(hf16, WRt, T, N);
    // attention + segment softmax
    k_attention<<<dim3((unsigned)(((long)E * 8 + 255) / 256)), b256, 0, stream>>>(
        T, rel, src, dst, ety, attbuf, nmax, N, E);
    k_expsum<<<dim3((unsigned)((E + 255) / 256)), b256, 0, stream>>>(
        attbuf, dst, nmax, wbuf, nsum, E);
    // layer 1
    k_aggregate<<<dim3((unsigned)(((long)E * 16 + 255) / 256)), b256, 0, stream>>>(
        ent, wbuf, nsum, src, dst, hnb, E);
    k_prod_f16<<<dim3(ndBlocks), b256, 0, stream>>>(ent, hnb, pf16, nd);
    k_layer<<<dim3(nTiles), dim3(128), 0, stream>>>(pf16, W0t, h1, N, 64);
    // layer 2
    k_zero<<<dim3(ndBlocks), b256, 0, stream>>>(hnb, nd);
    k_aggregate<<<dim3((unsigned)(((long)E * 16 + 255) / 256)), b256, 0, stream>>>(
        h1, wbuf, nsum, src, dst, hnb, E);
    k_prod_f16<<<dim3(ndBlocks), b256, 0, stream>>>(h1, hnb, pf16, nd);
    k_layer<<<dim3(nTiles), dim3(64), 0, stream>>>(pf16, W1t, h2, N, 32);
    // concat + L2 normalize
    k_output<<<dim3((unsigned)((N + 7) / 8)), b256, 0, stream>>>(ent, h1, h2, out, N);
}
